// RotaryAttention_16286515987065
// MI455X (gfx1250) — matrix operand order repaired and checked by reading, not compiled, not run on hardware
//
#include <hip/hip_runtime.h>

// ---------------------------------------------------------------------------
// RotaryAttention for MI455X (gfx1250), wave32 + v_wmma_f32_16x16x32_bf16.
// Compute-bound (~0.41 TFLOP vs ~200MB traffic at 23.3TB/s): all matmuls run
// on the bf16 WMMA path with fp32 accumulation.
//  * GEMMs: 32x64 register macro-tile per wave (2M x 4N WMMA tiles) -> each
//    A fragment feeds 4 WMMAs, each B fragment 2, and 8 independent
//    accumulators cover the WMMA->WMMA hazard window without NOPs.
//  * Weights pre-transposed to [N,K] bf16 so every B fragment is two
//    contiguous 16B/lane loads; V stored transposed ([b,h,hd,s]) so the P*V
//    B-fragments are contiguous too.
//  * RoPE fused into the QKV epilogue via lane __shfl_xor (pair partner is
//    the adjacent N lane in the WMMA C layout).
//  * Flash attention, online softmax; S->P operand transpose through LDS
//    guarded by an explicit s_wait_dscnt (CDNA5 split counters).
//  * global_prefetch_b8 with near-scope locality for the next K-step.
// ---------------------------------------------------------------------------

#define B_    4
#define S_    2048
#define D_    2048
#define H_    16
#define HD_   128
#define M_    (B_ * S_)     // 8192 rows
#define NQKV_ (3 * D_)      // 6144

typedef __attribute__((ext_vector_type(16))) __bf16 v16bf;
typedef __attribute__((ext_vector_type(8)))  float  v8f;

struct U4 { unsigned int x, y, z, w; };
struct F4 { float x, y, z, w; };

union FragBF {
  v16bf v;
  U4 q[2];
  unsigned short h[16];
};

__device__ __forceinline__ unsigned short f2bf(float f) {
  unsigned int u = __float_as_uint(f);
  u += 0x7FFFu + ((u >> 16) & 1u);          // round-to-nearest-even
  return (unsigned short)(u >> 16);
}

__device__ __forceinline__ v8f wmma_bf16(const FragBF& a, const FragBF& b, v8f c) {
  return __builtin_amdgcn_wmma_f32_16x16x32_bf16(
      /*neg_a=*/false, a.v, /*neg_b=*/false, b.v,
      /*c_mod=*/(short)0, c, /*reuse_a=*/false, /*reuse_b=*/false);
}

__device__ __forceinline__ void cvt8(const float* p, unsigned short* h) {
  F4 a = *(const F4*)p;
  F4 b = *(const F4*)(p + 4);
  h[0] = f2bf(a.x); h[1] = f2bf(a.y); h[2] = f2bf(a.z); h[3] = f2bf(a.w);
  h[4] = f2bf(b.x); h[5] = f2bf(b.y); h[6] = f2bf(b.z); h[7] = f2bf(b.w);
}

// A-fragment layout: lane<16 holds M=lane, K={0..7,16..23}; lane>=16 holds
// M=lane-16, K={8..15,24..31}. Pointer must already include (lane&15) row and
// hi*8 column offsets.
__device__ __forceinline__ void load_afrag_f32(FragBF& f, const float* p) {
  cvt8(p, f.h);
  cvt8(p + 16, f.h + 8);
}
__device__ __forceinline__ void load_afrag_bf16(FragBF& f, const unsigned short* p) {
  f.q[0] = *(const U4*)p;
  f.q[1] = *(const U4*)(p + 16);
}
// B-fragment layout: lane<16 holds N=lane K={0..7,16..23}, lane>=16 holds
// N=lane-16 K={8..15,24..31}. Pointer must include hi*8 offset.
__device__ __forceinline__ void load_bfrag(FragBF& f, const unsigned short* p) {
  f.q[0] = *(const U4*)p;
  f.q[1] = *(const U4*)(p + 16);
}

// --------------------------- weight transpose ------------------------------
// in:  fp32 [Kdim, Ndim] row-major   out: bf16 [Ndim, Kdim] row-major
__global__ void k_w_transpose_bf16(const float* __restrict__ in,
                                   unsigned short* __restrict__ out,
                                   int Kdim, int Ndim) {
  long long idx = (long long)blockIdx.x * blockDim.x + threadIdx.x;
  long long total = (long long)Kdim * Ndim;
  if (idx >= total) return;
  int k = (int)(idx / Ndim);
  int n = (int)(idx - (long long)k * Ndim);
  out[(long long)n * Kdim + k] = f2bf(in[idx]);
}

// ------------------------ QKV GEMM + bias + RoPE ---------------------------
// x[M_,D_] fp32 * WqkvT[NQKV_,D_] bf16; 32x64 tile per wave; epilogue routes
// each element to Q/K (RoPE'd, [b,h,s,hd] bf16) or Vt ([b,h,hd,s] bf16).
__global__ void __launch_bounds__(256)
k_qkv_rope(const float* __restrict__ x, const unsigned short* __restrict__ WqkvT,
           const float* __restrict__ bias, unsigned short* __restrict__ Qb,
           unsigned short* __restrict__ Kb, unsigned short* __restrict__ Vt) {
  const int lane  = threadIdx.x & 31;
  const int wave  = threadIdx.x >> 5;
  const int lhalf = lane & 15;
  const int hi    = lane >> 4;
  const int NTN   = NQKV_ / 64;             // 96 macro-tiles along N
  const int tile  = blockIdx.x * 8 + wave;  // 256*96 total, exact
  const int mt = tile / NTN, nt = tile % NTN;
  const int m0 = mt * 32, n0 = nt * 64;

  const float* ap0 = x + (long long)(m0 + lhalf) * D_ + hi * 8;
  const float* ap1 = ap0 + 16 * D_;
  const unsigned short* bp[4];
#pragma unroll
  for (int ni = 0; ni < 4; ++ni)
    bp[ni] = WqkvT + (long long)(n0 + ni * 16 + lhalf) * D_ + hi * 8;

  v8f acc[2][4] = {};
  for (int kk = 0; kk < D_; kk += 32) {
    FragBF a0, a1;
    load_afrag_f32(a0, ap0 + kk);
    load_afrag_f32(a1, ap1 + kk);
    if (kk + 32 < D_) {
      __builtin_prefetch(ap0 + kk + 32, 0, 3);   // near-scope global_prefetch_b8
      __builtin_prefetch(ap1 + kk + 32, 0, 3);
    }
#pragma unroll
    for (int ni = 0; ni < 4; ++ni) {
      FragBF b;
      load_bfrag(b, bp[ni] + kk);
      if (kk + 32 < D_) __builtin_prefetch(bp[ni] + kk + 32, 0, 3);
      acc[0][ni] = wmma_bf16(a0, b, acc[0][ni]);
      acc[1][ni] = wmma_bf16(a1, b, acc[1][ni]);
    }
  }

  // Epilogue: bias + RoPE + scatter. Each 16-wide N subtile stays inside one
  // q/k/v part (2048-aligned) and one head (128-aligned).
#pragma unroll
  for (int ni = 0; ni < 4; ++ni) {
    const int col  = n0 + ni * 16 + lhalf;
    const float bv = bias[col];
    const int part = col >> 11;        // 0=q 1=k 2=v
    const int dcol = col & (D_ - 1);
    const int head = dcol >> 7;
    const int hdim = dcol & (HD_ - 1);
    // inv_freq = 10000^(-(2*(hdim/2))/128)
    const float inv_freq =
        __expf(-(float)(hdim & ~1) * (9.210340371976184f / 128.0f));
#pragma unroll
    for (int mi = 0; mi < 2; ++mi) {
#pragma unroll
      for (int r = 0; r < 8; ++r) {
        const int grow  = m0 + mi * 16 + hi * 8 + r;  // [0, 8192)
        const int pos   = grow & (S_ - 1);
        const int batch = grow >> 11;
        float v = acc[mi][ni][r] + bv;
        float partner = __shfl_xor(v, 1, 32);  // RoPE pair partner
        if (part == 2) {
          Vt[((long long)(batch * H_ + head) * HD_ + hdim) * S_ + pos] = f2bf(v);
        } else {
          float ang = (float)pos * inv_freq;
          float cs = __cosf(ang), sn = __sinf(ang);
          float rot = (hdim & 1) ? fmaf(v, cs, partner * sn)
                                 : fmaf(v, cs, -partner * sn);
          long long off = ((long long)(batch * H_ + head) * S_ + pos) * HD_ + hdim;
          if (part == 0) Qb[off] = f2bf(rot);
          else           Kb[off] = f2bf(rot);
        }
      }
    }
  }
}

// --------------------------- flash attention -------------------------------
// One wave per (b,h, 16-query tile). Online softmax over 32-key blocks.
__global__ void __launch_bounds__(32)
k_attn(const unsigned short* __restrict__ Qb, const unsigned short* __restrict__ Kb,
       const unsigned short* __restrict__ Vt, unsigned short* __restrict__ ctx) {
  __shared__ __align__(16) unsigned short Pld[16 * 32];  // P tile staging
  const int lane  = threadIdx.x & 31;
  const int lhalf = lane & 15;
  const int hi    = lane >> 4;
  const int q0    = blockIdx.x * 16;
  const int bh    = blockIdx.y;
  const int batch = bh >> 4;
  const int head  = bh & 15;
  const float scale = 0.08838834764831845f;  // 1/sqrt(128)

  FragBF qf[4];
  const unsigned short* qbase = Qb + ((long long)bh * S_ + q0 + lhalf) * HD_ + hi * 8;
#pragma unroll
  for (int ks = 0; ks < 4; ++ks)
    load_afrag_bf16(qf[ks], qbase + ks * 32);

  float m[8], lsum[8];
  v8f o[8] = {};
#pragma unroll
  for (int r = 0; r < 8; ++r) { m[r] = -3.0e38f; lsum[r] = 0.0f; }

  const int kbmax = (q0 + 15) >> 5;
  for (int kb = 0; kb <= kbmax; ++kb) {
    const int c0 = kb * 32;
    v8f s0 = {}, s1 = {};
#pragma unroll
    for (int ks = 0; ks < 4; ++ks) {
      FragBF b0, b1;
      const unsigned short* kp0 =
          Kb + ((long long)bh * S_ + c0 + lhalf) * HD_ + ks * 32 + hi * 8;
      load_bfrag(b0, kp0);
      load_bfrag(b1, kp0 + 16 * HD_);
      s0 = wmma_bf16(qf[ks], b0, s0);
      s1 = wmma_bf16(qf[ks], b1, s1);
    }

    float p0a[8], p1a[8], alpha[8];
#pragma unroll
    for (int r = 0; r < 8; ++r) {
      const int qi   = q0 + hi * 8 + r;
      const int kidx = c0 + lhalf;
      float a0 = (kidx      <= qi) ? s0[r] * scale : -3.0e38f;
      float a1 = (kidx + 16 <= qi) ? s1[r] * scale : -3.0e38f;
      float t = fmaxf(a0, a1);
      t = fmaxf(t, __shfl_xor(t, 1, 32));
      t = fmaxf(t, __shfl_xor(t, 2, 32));
      t = fmaxf(t, __shfl_xor(t, 4, 32));
      t = fmaxf(t, __shfl_xor(t, 8, 32));
      float mn = fmaxf(m[r], t);
      alpha[r] = __expf(m[r] - mn);
      m[r] = mn;
      float p0 = __expf(a0 - mn);
      float p1 = __expf(a1 - mn);
      float rs = p0 + p1;
      rs += __shfl_xor(rs, 1, 32);
      rs += __shfl_xor(rs, 2, 32);
      rs += __shfl_xor(rs, 4, 32);
      rs += __shfl_xor(rs, 8, 32);
      lsum[r] = lsum[r] * alpha[r] + rs;
      p0a[r] = p0; p1a[r] = p1;
    }
#pragma unroll
    for (int nt = 0; nt < 8; ++nt)
#pragma unroll
      for (int r = 0; r < 8; ++r)
        o[nt][r] *= alpha[r];

    // Transpose P from C layout to A layout through LDS
#pragma unroll
    for (int r = 0; r < 8; ++r) {
      const int row = hi * 8 + r;
      Pld[row * 32 + lhalf]      = f2bf(p0a[r]);
      Pld[row * 32 + 16 + lhalf] = f2bf(p1a[r]);
    }
    asm volatile("s_wait_dscnt 0" ::: "memory");   // CDNA5 split DS counter
    FragBF pf;
    pf.q[0] = *(const U4*)(&Pld[lhalf * 32 + hi * 8]);
    pf.q[1] = *(const U4*)(&Pld[lhalf * 32 + hi * 8 + 16]);

#pragma unroll
    for (int nt = 0; nt < 8; ++nt) {
      FragBF vf;
      load_bfrag(vf, Vt + ((long long)bh * HD_ + nt * 16 + lhalf) * S_ + c0 + hi * 8);
      o[nt] = wmma_bf16(pf, vf, o[nt]);
    }
  }

  // normalize and store ctx as [b, s, h*hd] bf16 (= reshape(b,s,d))
#pragma unroll
  for (int nt = 0; nt < 8; ++nt) {
#pragma unroll
    for (int r = 0; r < 8; ++r) {
      const int row = q0 + hi * 8 + r;
      const int col = head * HD_ + nt * 16 + lhalf;
      float val = o[nt][r] / lsum[r];
      ctx[((long long)batch * S_ + row) * D_ + col] = f2bf(val);
    }
  }
}

// ------------------------- output projection -------------------------------
// ctx[M_,D_] bf16 * WoutT[D_,D_] bf16 + bias -> out fp32; 32x64 tile per wave.
__global__ void __launch_bounds__(256)
k_out_gemm(const unsigned short* __restrict__ ctx,
           const unsigned short* __restrict__ WoutT,
           const float* __restrict__ bias, float* __restrict__ out) {
  const int lane  = threadIdx.x & 31;
  const int wave  = threadIdx.x >> 5;
  const int lhalf = lane & 15;
  const int hi    = lane >> 4;
  const int NTN   = D_ / 64;                // 32 macro-tiles along N
  const int tile  = blockIdx.x * 8 + wave;  // 256*32 total, exact
  const int mt = tile / NTN, nt = tile % NTN;
  const int m0 = mt * 32, n0 = nt * 64;

  const unsigned short* ap0 = ctx + (long long)(m0 + lhalf) * D_ + hi * 8;
  const unsigned short* ap1 = ap0 + 16 * D_;
  const unsigned short* bp[4];
#pragma unroll
  for (int ni = 0; ni < 4; ++ni)
    bp[ni] = WoutT + (long long)(n0 + ni * 16 + lhalf) * D_ + hi * 8;

  v8f acc[2][4] = {};
  for (int kk = 0; kk < D_; kk += 32) {
    FragBF a0, a1;
    load_afrag_bf16(a0, ap0 + kk);
    load_afrag_bf16(a1, ap1 + kk);
    if (kk + 32 < D_) {
      __builtin_prefetch(ap0 + kk + 32, 0, 3);
      __builtin_prefetch(ap1 + kk + 32, 0, 3);
    }
#pragma unroll
    for (int ni = 0; ni < 4; ++ni) {
      FragBF b;
      load_bfrag(b, bp[ni] + kk);
      if (kk + 32 < D_) __builtin_prefetch(bp[ni] + kk + 32, 0, 3);
      acc[0][ni] = wmma_bf16(a0, b, acc[0][ni]);
      acc[1][ni] = wmma_bf16(a1, b, acc[1][ni]);
    }
  }

#pragma unroll
  for (int ni = 0; ni < 4; ++ni) {
    const int col  = n0 + ni * 16 + lhalf;
    const float bv = bias[col];
#pragma unroll
    for (int mi = 0; mi < 2; ++mi) {
#pragma unroll
      for (int r = 0; r < 8; ++r) {
        const int row = m0 + mi * 16 + hi * 8 + r;
        out[(long long)row * D_ + col] = acc[mi][ni][r] + bv;
      }
    }
  }
}

// ------------------------------- launcher ----------------------------------
extern "C" void kernel_launch(void* const* d_in, const int* in_sizes, int n_in,
                              void* d_out, int out_size, void* d_ws, size_t ws_size,
                              hipStream_t stream) {
  (void)in_sizes; (void)n_in; (void)out_size; (void)ws_size;
  const float* x       = (const float*)d_in[0];
  // d_in[1] = attn_mask (causal; implemented analytically), d_in[6] = num_heads(=16)
  const float* Wqkv_w  = (const float*)d_in[2];
  const float* Wqkv_b  = (const float*)d_in[3];
  const float* Wout_w  = (const float*)d_in[4];
  const float* Wout_b  = (const float*)d_in[5];
  float* out = (float*)d_out;

  // workspace carve-up (bf16 stored as raw u16); total = 160 MiB
  char* ws = (char*)d_ws;
  unsigned short* WqkvT = (unsigned short*)ws; ws += (size_t)NQKV_ * D_ * 2;
  unsigned short* WoutT = (unsigned short*)ws; ws += (size_t)D_ * D_ * 2;
  unsigned short* Qb    = (unsigned short*)ws; ws += (size_t)B_ * H_ * S_ * HD_ * 2;
  unsigned short* Kb    = (unsigned short*)ws; ws += (size_t)B_ * H_ * S_ * HD_ * 2;
  unsigned short* Vt    = (unsigned short*)ws; ws += (size_t)B_ * H_ * S_ * HD_ * 2;
  unsigned short* ctx   = (unsigned short*)ws; ws += (size_t)M_ * D_ * 2;

  {
    long long n = (long long)D_ * NQKV_;
    k_w_transpose_bf16<<<(unsigned)((n + 255) / 256), 256, 0, stream>>>(
        Wqkv_w, WqkvT, D_, NQKV_);
  }
  {
    long long n = (long long)D_ * D_;
    k_w_transpose_bf16<<<(unsigned)((n + 255) / 256), 256, 0, stream>>>(
        Wout_w, WoutT, D_, D_);
  }
  k_qkv_rope<<<(M_ / 32) * (NQKV_ / 64) / 8, 256, 0, stream>>>(
      x, WqkvT, Wqkv_b, Qb, Kb, Vt);
  k_attn<<<dim3(S_ / 16, B_ * H_), 32, 0, stream>>>(Qb, Kb, Vt, ctx);
  k_out_gemm<<<(M_ / 32) * (D_ / 64) / 8, 256, 0, stream>>>(
      ctx, WoutT, Wout_b, out);
}
